// Transformer_6751688589730
// MI455X (gfx1250) — matrix operand order repaired and checked by reading, compiled, not run on hardware
//
#include <hip/hip_runtime.h>
#include <hip/hip_bf16.h>

// ---------------------------------------------------------------------------
// GPT-2 style transformer forward for gfx1250 (MI455X), wave32 + WMMA f16.
// Compute-bound (~455 GFLOP vs ~406MB weight traffic @ 23.3TB/s), so all
// GEMMs run on v_wmma_f32_16x16x32_f16 with f32 accumulate; f32 weights are
// converted to f16 while staging to LDS. Straight f16 copies (A tiles,
// attention K tiles) use the CDNA5 async global->LDS pipe (ASYNCcnt).
// ---------------------------------------------------------------------------

typedef __attribute__((ext_vector_type(16))) _Float16 v16h;
typedef __attribute__((ext_vector_type(8)))  _Float16 v8h;
typedef __attribute__((ext_vector_type(2)))  _Float16 v2h;
typedef __attribute__((ext_vector_type(8)))  float    v8f;

union Frag16 { v16h v; v8h h[2]; };

#define TILE 128
#define KT   32
#define LDA  40   // LDS row stride (halves): 32 + 8 pad; 80B = 5*16 keeps b128 align
#define LDB  40

// ---- CDNA5 async global->LDS copy (GLOBAL_LOAD_ASYNC_TO_LDS_B128) ---------
// Generic pointers to __shared__ carry the LDS byte offset in their low 32
// bits (flat-aperture mapping), which is exactly the per-lane VDST operand.
__device__ __forceinline__ void async_ld_b128(void* lds, const void* g)
{
    unsigned l = (unsigned)(unsigned long long)lds;
    asm volatile("global_load_async_to_lds_b128 %0, %1, off"
                 :: "v"(l), "v"((unsigned long long)g) : "memory");
}
__device__ __forceinline__ void wait_async0()
{
    asm volatile("s_wait_asynccnt 0" ::: "memory");
}

// ---------------------------------------------------------------------------
// Embedding: x[t][d] = emb[tokens[t]][d] + pos_emb[t % M][d]
// ---------------------------------------------------------------------------
__global__ __launch_bounds__(256)
void embed_k(const int* __restrict__ tokens, const float* __restrict__ emb,
             const float* __restrict__ pos, float* __restrict__ x)
{
    const int t = blockIdx.x;            // 0..2047
    const int m = t & 1023;
    const int tok = tokens[t];
    const float* e = emb + (size_t)tok * 1024;
    const float* p = pos + (size_t)m * 1024;
    float* xr = x + (size_t)t * 1024;
    for (int i = threadIdx.x; i < 1024; i += 256)
        xr[i] = e[i] + p[i];
}

// ---------------------------------------------------------------------------
// LayerNorm (biased var, eps=1e-5) -> f16 output for WMMA consumption.
// ---------------------------------------------------------------------------
__global__ __launch_bounds__(256)
void layernorm_f16(const float* __restrict__ x, const float* __restrict__ s,
                   const float* __restrict__ b, _Float16* __restrict__ out)
{
    const int t = blockIdx.x;
    const int tid = threadIdx.x;
    const float* xr = x + (size_t)t * 1024;
    __shared__ float red[256];

    float ls = 0.f;
    for (int i = tid; i < 1024; i += 256) ls += xr[i];
    red[tid] = ls; __syncthreads();
    for (int o = 128; o > 0; o >>= 1) {
        if (tid < o) red[tid] += red[tid + o];
        __syncthreads();
    }
    const float mu = red[0] * (1.0f / 1024.0f);
    __syncthreads();

    float lv = 0.f;
    for (int i = tid; i < 1024; i += 256) { float d = xr[i] - mu; lv += d * d; }
    red[tid] = lv; __syncthreads();
    for (int o = 128; o > 0; o >>= 1) {
        if (tid < o) red[tid] += red[tid + o];
        __syncthreads();
    }
    const float rstd = rsqrtf(red[0] * (1.0f / 1024.0f) + 1e-5f);

    _Float16* orow = out + (size_t)t * 1024;
    for (int i = tid; i < 1024; i += 256)
        orow[i] = (_Float16)((xr[i] - mu) * rstd * s[i] + b[i]);
}

// ---------------------------------------------------------------------------
// Tiled WMMA GEMM: C[rows,cols] = epilogue(Af16[rows,kdim] @ Wf32[kdim,cols])
// Block: 256 thr = 8 waves; tile 128x128; wave tile 32x64 (2x4 of 16x16).
// EP: 0 = f16 out (QKV) | 1 = bias+GELU f16 (W1)
//     2 = (+bias)+residual into f32 x (Po,W2) | 3 = bias, f32 out (head)
// ---------------------------------------------------------------------------
template<int EP>
__global__ __launch_bounds__(256)
void gemm_wmma(const _Float16* __restrict__ A, const float* __restrict__ W,
               const float* __restrict__ bias, float* __restrict__ resid,
               void* __restrict__ out, int rows, int cols, int kdim)
{
    __shared__ __align__(16) _Float16 sA[TILE][LDA];   // [row][k]
    __shared__ __align__(16) _Float16 sB[TILE][LDB];   // [col][k] (transposed)

    const int tid  = threadIdx.x;
    const int lane = tid & 31;
    const int wave = tid >> 5;        // 0..7
    const int wr   = wave >> 1;       // row group (32 rows)
    const int wc   = wave & 1;        // col group (64 cols)
    const int g    = lane >> 4;       // lane half-group
    const int ln   = lane & 15;
    const int row0 = blockIdx.y * TILE;
    const int col0 = blockIdx.x * TILE;

    v8f acc[2][4] = {};

    for (int kt = 0; kt < kdim; kt += KT) {
        // ---- stage A tile via async global->LDS b128 (2 per thread) ------
#pragma unroll
        for (int s2 = 0; s2 < 2; ++s2) {
            int seg = tid * 2 + s2;            // 0..511
            int r   = seg >> 2;                // 0..127
            int c   = (seg & 3) * 8;           // 0,8,16,24
            async_ld_b128(&sA[r][c],
                          A + (size_t)(row0 + r) * kdim + kt + c);
        }
        // ---- stage W tile transposed + f32->f16, packed b32 stores -------
        {
            int kk = (tid & 15) * 2;           // 0,2,..,30
            int cg = (tid >> 4) * 8;           // 0,8,..,120
            const float* w0 = W + (size_t)(kt + kk) * cols + col0 + cg;
            const float* w1 = w0 + cols;
            if (col0 + TILE <= cols) {         // fast path: no edge guard
#pragma unroll
                for (int c = 0; c < 8; ++c) {
                    v2h pk = { (_Float16)w0[c], (_Float16)w1[c] };
                    *(v2h*)&sB[cg + c][kk] = pk;
                }
            } else {
#pragma unroll
                for (int c = 0; c < 8; ++c) {
                    int col = col0 + cg + c;
                    float a0 = (col < cols) ? w0[c] : 0.0f;
                    float a1 = (col < cols) ? w1[c] : 0.0f;
                    v2h pk = { (_Float16)a0, (_Float16)a1 };
                    *(v2h*)&sB[cg + c][kk] = pk;
                }
            }
            // prefetch next K-tile of W into cache (global_prefetch_b8)
            if (kt + KT < kdim)
                __builtin_prefetch(W + (size_t)(kt + KT + kk) * cols + col0 + cg, 0, 1);
        }
        wait_async0();
        __syncthreads();

        // ---- fragments (documented gfx1250 16-bit A/B VGPR layouts) ------
        Frag16 af[2], bf[4];
#pragma unroll
        for (int mt = 0; mt < 2; ++mt) {
            int r = wr * 32 + mt * 16 + ln;                 // A: lane m = l%16
            af[mt].h[0] = *(const v8h*)&sA[r][g * 8];       // k = g*8 .. +7
            af[mt].h[1] = *(const v8h*)&sA[r][16 + g * 8];  // k = 16+g*8 .. +7
        }
#pragma unroll
        for (int nt = 0; nt < 4; ++nt) {
            int c = wc * 64 + nt * 16 + ln;                 // B: lane n = l%16
            bf[nt].h[0] = *(const v8h*)&sB[c][g * 8];       // k = g*8 .. +7
            bf[nt].h[1] = *(const v8h*)&sB[c][16 + g * 8];  // k = 16+g*8 .. +7
        }
#pragma unroll
        for (int mt = 0; mt < 2; ++mt)
#pragma unroll
            for (int nt = 0; nt < 4; ++nt)
                acc[mt][nt] = __builtin_amdgcn_wmma_f32_16x16x32_f16(
                    false, af[mt].v, false, bf[nt].v, (short)0, acc[mt][nt],
                    false, false);

        __syncthreads();
    }

    // ---- epilogue: C layout lane -> row = r + 8*g, col = l%16 ------------
#pragma unroll
    for (int mt = 0; mt < 2; ++mt) {
#pragma unroll
        for (int nt = 0; nt < 4; ++nt) {
#pragma unroll
            for (int r = 0; r < 8; ++r) {
                int row = row0 + wr * 32 + mt * 16 + r + 8 * g;
                int col = col0 + wc * 64 + nt * 16 + ln;
                if (col >= cols) continue;
                float v = acc[mt][nt][r];
                if (bias) v += bias[col];
                size_t idx = (size_t)row * cols + col;
                if (EP == 0) {
                    ((_Float16*)out)[idx] = (_Float16)v;
                } else if (EP == 1) {
                    float u = v;
                    float t3 = u + 0.044715f * u * u * u;
                    float gg = 0.5f * u * (1.0f + tanhf(0.7978845608028654f * t3));
                    ((_Float16*)out)[idx] = (_Float16)gg;
                } else if (EP == 2) {
                    resid[idx] = resid[idx] + v;
                } else {
                    ((float*)out)[idx] = v;
                }
            }
        }
    }
}

// ---------------------------------------------------------------------------
// Flash attention, causal, H=16, K=64, M=1024.
// Grid (M/64, H, B); block 128 = 4 waves; wave owns 16 query rows.
// qkv row layout: [h*192 + {q:0..63, k:64..127, v:128..191}], f16.
// ---------------------------------------------------------------------------
__global__ __launch_bounds__(128)
void flash_attn(const _Float16* __restrict__ qkv, _Float16* __restrict__ o)
{
    const int R    = blockIdx.x * 64;
    const int h    = blockIdx.y;
    const int b    = blockIdx.z;
    const int tid  = threadIdx.x;
    const int lane = tid & 31;
    const int wave = tid >> 5;          // 0..3
    const int g    = lane >> 4;
    const int ln   = lane & 15;
    const int r0   = R + wave * 16;     // wave's first query row (in batch)
    const size_t base = (size_t)b * 1024 * 3072;

    __shared__ __align__(16) _Float16 sK[32][72];       // [key][kdim], 144B rows
    __shared__ __align__(16) _Float16 sV[64][40];       // [vdim][key] (transposed)
    __shared__ __align__(16) _Float16 sP[4][16][40];    // per-wave P tile

    // Q fragments: A 16x64 = two 16x32 chunks, loaded straight from global
    Frag16 qf[2];
    {
        const _Float16* qrow = qkv + base + (size_t)(r0 + ln) * 3072 + h * 192;
#pragma unroll
        for (int c = 0; c < 2; ++c) {
            qf[c].h[0] = *(const v8h*)(qrow + c * 32 + g * 8);
            qf[c].h[1] = *(const v8h*)(qrow + c * 32 + 16 + g * 8);
        }
    }

    v8f  oacc[4] = {};
    float mrow[8], lrow[8];
#pragma unroll
    for (int r = 0; r < 8; ++r) { mrow[r] = -1e30f; lrow[r] = 0.f; }

    for (int j0 = 0; j0 < R + 64; j0 += 32) {
        // ---- K block staged via async global->LDS b128 -------------------
#pragma unroll
        for (int s2 = 0; s2 < 2; ++s2) {
            int seg = tid * 2 + s2;          // 0..255
            int key = seg >> 3;              // 0..31
            int c   = (seg & 7) * 8;         // 0..56
            async_ld_b128(&sK[key][c],
                qkv + base + (size_t)(j0 + key) * 3072 + h * 192 + 64 + c);
        }
        // ---- V block staged transposed: sV[vdim][key] --------------------
        {
            int key  = tid >> 2;             // 0..31
            int vseg = (tid & 3) * 16;       // 0,16,32,48
            const _Float16* vp =
                qkv + base + (size_t)(j0 + key) * 3072 + h * 192 + 128 + vseg;
            v8h a0 = *(const v8h*)vp;
            v8h a1 = *(const v8h*)(vp + 8);
#pragma unroll
            for (int i = 0; i < 8; ++i) {
                sV[vseg + i][key]     = a0[i];
                sV[vseg + 8 + i][key] = a1[i];
            }
        }
        wait_async0();
        __syncthreads();

        if (j0 <= r0 + 15) {   // skip fully-masked key blocks for this wave
            // ---- S = Q @ K^T : 16x32 logits = 2 C tiles ------------------
            v8f sc[2] = {};
#pragma unroll
            for (int st = 0; st < 2; ++st) {
                int key = st * 16 + ln;
#pragma unroll
                for (int ch = 0; ch < 2; ++ch) {
                    Frag16 kf;
                    kf.h[0] = *(const v8h*)&sK[key][ch * 32 + g * 8];
                    kf.h[1] = *(const v8h*)&sK[key][ch * 32 + 16 + g * 8];
                    sc[st] = __builtin_amdgcn_wmma_f32_16x16x32_f16(
                        false, qf[ch].v, false, kf.v, (short)0, sc[st],
                        false, false);
                }
            }
            // ---- mask + online softmax (shfl_xor row reduce, wave32) -----
#pragma unroll
            for (int r = 0; r < 8; ++r) {
                int row = r0 + r + 8 * g;
                float e0 = sc[0][r] * 0.125f;
                float e1 = sc[1][r] * 0.125f;
                if (j0 + ln      > row) e0 = -1e30f;
                if (j0 + 16 + ln > row) e1 = -1e30f;
                float mx = fmaxf(e0, e1);
#pragma unroll
                for (int d = 8; d >= 1; d >>= 1) mx = fmaxf(mx, __shfl_xor(mx, d, 32));
                float mnew  = fmaxf(mrow[r], mx);
                float alpha = __expf(mrow[r] - mnew);
                float p0 = __expf(e0 - mnew);
                float p1 = __expf(e1 - mnew);
                float sum = p0 + p1;
#pragma unroll
                for (int d = 8; d >= 1; d >>= 1) sum += __shfl_xor(sum, d, 32);
                lrow[r] = lrow[r] * alpha + sum;
                mrow[r] = mnew;
#pragma unroll
                for (int nt = 0; nt < 4; ++nt) oacc[nt][r] *= alpha;
                // spill P to per-wave LDS tile (re-layout C -> A fragment)
                int m = r + 8 * g;
                sP[wave][m][ln]      = (_Float16)p0;
                sP[wave][m][16 + ln] = (_Float16)p1;
            }
            // ---- O += P @ V  (A 16x32 from sP, B 32x16 from sV) ----------
            Frag16 pf;
            pf.h[0] = *(const v8h*)&sP[wave][ln][g * 8];
            pf.h[1] = *(const v8h*)&sP[wave][ln][16 + g * 8];
#pragma unroll
            for (int nt = 0; nt < 4; ++nt) {
                Frag16 vf;
                vf.h[0] = *(const v8h*)&sV[nt * 16 + ln][g * 8];
                vf.h[1] = *(const v8h*)&sV[nt * 16 + ln][16 + g * 8];
                oacc[nt] = __builtin_amdgcn_wmma_f32_16x16x32_f16(
                    false, pf.v, false, vf.v, (short)0, oacc[nt], false, false);
            }
        }
        __syncthreads();
    }

    // ---- normalize and store concat-head output (f16) --------------------
#pragma unroll
    for (int nt = 0; nt < 4; ++nt) {
#pragma unroll
        for (int r = 0; r < 8; ++r) {
            int row = r0 + r + 8 * g;
            int col = h * 64 + nt * 16 + ln;
            float inv = 1.0f / lrow[r];
            o[((size_t)b * 1024 + row) * 1024 + col] = (_Float16)(oacc[nt][r] * inv);
        }
    }
}

// ---------------------------------------------------------------------------
// Host-side orchestration
// ---------------------------------------------------------------------------
extern "C" void kernel_launch(void* const* d_in, const int* in_sizes, int n_in,
                              void* d_out, int out_size, void* d_ws, size_t ws_size,
                              hipStream_t stream) {
    (void)in_sizes; (void)n_in; (void)out_size; (void)ws_size;
    const int*   tokens = (const int*)  d_in[0];
    const float* emb    = (const float*)d_in[1];
    const float* pos    = (const float*)d_in[2];
    const float* ln1_s  = (const float*)d_in[3];
    const float* ln1_b  = (const float*)d_in[4];
    const float* Pi     = (const float*)d_in[5];
    const float* Po     = (const float*)d_in[6];
    const float* ln2_s  = (const float*)d_in[7];
    const float* ln2_b  = (const float*)d_in[8];
    const float* W1     = (const float*)d_in[9];
    const float* b1     = (const float*)d_in[10];
    const float* W2     = (const float*)d_in[11];
    const float* b2     = (const float*)d_in[12];
    const float* lnf_s  = (const float*)d_in[13];
    const float* lnf_b  = (const float*)d_in[14];
    const float* Wh     = (const float*)d_in[15];
    const float* bh     = (const float*)d_in[16];

    const int rows = 2048, D = 1024, L = 4, V = 50257;

    char* ws = (char*)d_ws;
    float*     x    = (float*)     ws;                       //  8 MB f32 [2048,1024]
    _Float16*  xn   = (_Float16*) (ws + (  8u << 20));       //  4 MB f16 [2048,1024]
    _Float16*  qkvb = (_Float16*) (ws + ( 12u << 20));       // 12 MB f16 [2048,3072]
    _Float16*  attn = (_Float16*) (ws + ( 24u << 20));       //  4 MB f16 [2048,1024]
    _Float16*  hbuf = (_Float16*) (ws + ( 28u << 20));       // 16 MB f16 [2048,4096]

    embed_k<<<rows, 256, 0, stream>>>(tokens, emb, pos, x);

    for (int l = 0; l < L; ++l) {
        layernorm_f16<<<rows, 256, 0, stream>>>(x, ln1_s + l * D, ln1_b + l * D, xn);
        gemm_wmma<0><<<dim3(3072 / 128, rows / 128), 256, 0, stream>>>(
            xn, Pi + (size_t)l * D * 3072, nullptr, nullptr, (void*)qkvb,
            rows, 3072, D);
        flash_attn<<<dim3(1024 / 64, 16, 2), 128, 0, stream>>>(qkvb, attn);
        gemm_wmma<2><<<dim3(1024 / 128, rows / 128), 256, 0, stream>>>(
            attn, Po + (size_t)l * D * D, nullptr, x, nullptr,
            rows, 1024, D);
        layernorm_f16<<<rows, 256, 0, stream>>>(x, ln2_s + l * D, ln2_b + l * D, xn);
        gemm_wmma<1><<<dim3(4096 / 128, rows / 128), 256, 0, stream>>>(
            xn, W1 + (size_t)l * D * 4096, b1 + (size_t)l * 4096, nullptr,
            (void*)hbuf, rows, 4096, D);
        gemm_wmma<2><<<dim3(1024 / 128, rows / 128), 256, 0, stream>>>(
            hbuf, W2 + (size_t)l * 4096 * D, b2 + l * D, x, nullptr,
            rows, 1024, 4096);
    }

    layernorm_f16<<<rows, 256, 0, stream>>>(x, lnf_s, lnf_b, xn);
    gemm_wmma<3><<<dim3((V + 127) / 128, rows / 128), 256, 0, stream>>>(
        xn, Wh, bh, nullptr, d_out, rows, V, D);
}
